// LSTMEncoder_54176717472336
// MI455X (gfx1250) — matrix operand order repaired and checked by reading, not compiled, not run on hardware
//
#include <hip/hip_runtime.h>

typedef __attribute__((ext_vector_type(16))) __bf16 v16bf;
typedef __attribute__((ext_vector_type(8)))  __bf16 v8bf;
typedef __attribute__((ext_vector_type(8)))  float  v8f;

#define B_   256
#define S_   512
#define D_   200
#define H_   256
#define KX   224          // x part padded: 200 real + 24 zero
#define KPAD 480          // KX + H_
#define KSTEPS 15         // KPAD / 32
#define NT_COUNT 64       // 1024 columns / 16
#define XH_STRIDE 488     // bf16 elements per LDS row (>=480, multiple of 8 for 16B align)

__device__ __forceinline__ __bf16 f2bf(float f) {
    unsigned u = __float_as_uint(f);
    unsigned r = (u + 0x7FFFu + ((u >> 16) & 1u)) >> 16;   // round-to-nearest-even
    unsigned short h = (unsigned short)r;
    return __builtin_bit_cast(__bf16, h);
}

// ---- fast activations on hardware TRANS ops (v_exp_f32 / v_rcp_f32 / v_tanh_f32) ----
__device__ __forceinline__ float fast_rcp(float x) {
#if __has_builtin(__builtin_amdgcn_rcpf)
    return __builtin_amdgcn_rcpf(x);
#else
    return 1.f / x;
#endif
}
__device__ __forceinline__ float fast_exp2(float x) {
#if __has_builtin(__builtin_amdgcn_exp2f)
    return __builtin_amdgcn_exp2f(x);
#else
    return __builtin_exp2f(x);
#endif
}
__device__ __forceinline__ float fast_sigmoid(float p) {
    // 1/(1+2^(-p*log2e))
    return fast_rcp(1.f + fast_exp2(p * -1.44269504088896340736f));
}
__device__ __forceinline__ float fast_tanh(float x) {
#if __has_builtin(__builtin_amdgcn_tanhf)
    return __builtin_amdgcn_tanhf(x);
#else
    // tanh(x) = 1 - 2/(2^(2x*log2e) + 1); saturates correctly at +/-inf of exp2
    float t = fast_exp2(x * 2.88539008177792681472f);
    return 1.f - 2.f * fast_rcp(t + 1.f);
#endif
}

// Pack the 4 gate weight matrices (each [456,256] f32, rows 0..199 = x-part,
// rows 200..455 = h-part) into WMMA-B-fragment-native bf16 layout:
//   Bfrag[nt][ks][lane][e]  =  Wpad[ks*32 + 8*(lane>>4) + (e<8 ? e : 8+e)][nt*16 + (lane&15)]
// where Wpad has 480 K-rows: [0,200)=Wx, [200,224)=0, [224,480)=Wh.
__global__ void pack_bfrag(const float* __restrict__ Wi, const float* __restrict__ Wf,
                           const float* __restrict__ Wo, const float* __restrict__ Wz,
                           __bf16* __restrict__ Bfrag) {
    int t = blockIdx.x * blockDim.x + threadIdx.x;
    if (t >= NT_COUNT * KSTEPS * 32) return;
    int nt   = t / (KSTEPS * 32);
    int rem  = t - nt * (KSTEPS * 32);
    int ks   = rem >> 5;
    int lane = rem & 31;
    int kb   = ks * 32 + (lane >> 4) * 8;   // padded K base for this lane half (table order)
    int g    = nt >> 4;                 // gate: 0=i,1=f,2=o,3=z
    int j0   = (nt & 15) * 16 + (lane & 15);   // column within gate
    const float* W = (g == 0) ? Wi : (g == 1) ? Wf : (g == 2) ? Wo : Wz;
    v16bf v;
    #pragma unroll
    for (int e = 0; e < 16; ++e) {
        int k = kb + (e < 8 ? e : 8 + e);
        int r; bool zero;
        if (k < D_)        { r = k;      zero = false; }
        else if (k < KX)   { r = 0;      zero = true;  }
        else               { r = k - (KX - D_); zero = false; }   // rows 200..455
        float x = zero ? 0.f : W[(size_t)r * H_ + j0];
        v[e] = f2bf(x);
    }
    *(v16bf*)(Bfrag + (size_t)t * 16) = v;
}

// Persistent LSTM: one workgroup per 16-batch tile, 8 waves.
// Wave w owns hidden columns [w*32, w*32+32): all 4 gates for that slice live in
// the wave's 8 accumulators, so the cell update is entirely in-register.
__global__ __launch_bounds__(256, 1)
void lstm_persistent(const float* __restrict__ x, const float* __restrict__ mask,
                     const float* __restrict__ bi_p, const float* __restrict__ bf_p,
                     const float* __restrict__ bo_p, const float* __restrict__ bz_p,
                     const __bf16* __restrict__ Bfrag, float* __restrict__ out) {
    __shared__ __align__(16) __bf16 xh[16 * XH_STRIDE];  // [m][480+pad] bf16
    __shared__ float lds_mask[16];

    const int tid  = threadIdx.x;
    const int wave = tid >> 5;
    const int lane = tid & 31;
    const int l16  = lane & 15;
    const int hi   = lane >> 4;
    const int b0   = blockIdx.x * 16;

    // per-lane bias registers: bias[g][h] for column j = wave*32 + h*16 + l16
    float bias[4][2];
    #pragma unroll
    for (int h = 0; h < 2; ++h) {
        int j = wave * 32 + h * 16 + l16;
        bias[0][h] = bi_p[j];
        bias[1][h] = bf_p[j];
        bias[2][h] = bo_p[j];
        bias[3][h] = bz_p[j];
    }

    float cst[16], hsum[16];                 // per-lane c and masked-h sum (m = 8*hi + r)
    #pragma unroll
    for (int q = 0; q < 16; ++q) { cst[q] = 0.f; hsum[q] = 0.f; }
    float smsum[8];
    #pragma unroll
    for (int r = 0; r < 8; ++r) smsum[r] = 0.f;

    // zero-init h region of the A matrix
    for (int idx = tid; idx < 16 * H_; idx += 256) {
        int m = idx >> 8, j = idx & 255;
        xh[m * XH_STRIDE + KX + j] = f2bf(0.f);
    }

    const __bf16* arow = xh + l16 * XH_STRIDE;

    for (int s = 0; s < S_; ++s) {
        // ---- stage x_t (bf16) + pad zeros + mask values ----
        for (int idx = tid; idx < 16 * KX; idx += 256) {
            int m = idx / KX, col = idx - m * KX;
            float v = (col < D_) ? x[((size_t)(b0 + m) * S_ + s) * D_ + col] : 0.f;
            xh[m * XH_STRIDE + col] = f2bf(v);
        }
        if (tid < 16) lds_mask[tid] = mask[(size_t)(b0 + tid) * S_ + s];
        __syncthreads();   // A: staging + previous h-writeback visible

        // consume mask now (barrier B below orders this vs next-step overwrite)
        float mk[8];
        #pragma unroll
        for (int r = 0; r < 8; ++r) { mk[r] = lds_mask[8 * hi + r]; smsum[r] += mk[r]; }

        // prefetch next step's x rows into cache while the GEMM runs
        if (s + 1 < S_) {
            int pm  = tid >> 4;            // row 0..15
            int pc  = (tid & 15) * 14;     // every 56B within the 800B row
            if (pc < D_)
                __builtin_prefetch(&x[((size_t)(b0 + pm) * S_ + (s + 1)) * D_ + pc], 0, 3);
        }

        // ---- GEMM: [16 x 480] @ [480 x 32-slice], 4 gates x 2 subtiles ----
        v8f acc[8];
        #pragma unroll
        for (int i = 0; i < 8; ++i) acc[i] = (v8f){0.f,0.f,0.f,0.f,0.f,0.f,0.f,0.f};

        for (int ks = 0; ks < KSTEPS; ++ks) {
            v8bf alo = *(const v8bf*)(arow + 32 * ks + hi * 8);
            v8bf ahh = *(const v8bf*)(arow + 32 * ks + 16 + hi * 8);
            v16bf a = __builtin_shufflevector(alo, ahh,
                        0,1,2,3,4,5,6,7,8,9,10,11,12,13,14,15);
            #pragma unroll
            for (int i = 0; i < 8; ++i) {
                int nt = (i >> 1) * 16 + wave * 2 + (i & 1);   // gate = i>>1, half = i&1
                const v16bf* bp = (const v16bf*)(Bfrag +
                        (((size_t)nt * KSTEPS + ks) * 32 + lane) * 16);
                acc[i] = __builtin_amdgcn_wmma_f32_16x16x32_bf16(
                            false, a, false, *bp, (short)0, acc[i], false, false);
            }
        }
        __syncthreads();   // B: all waves done reading xh before h-writeback

        // ---- in-register LSTM cell update (hardware TRANS activations) ----
        #pragma unroll
        for (int h = 0; h < 2; ++h) {
            int j = wave * 32 + h * 16 + l16;
            #pragma unroll
            for (int r = 0; r < 8; ++r) {
                int q = h * 8 + r;
                float pi = acc[0 * 2 + h][r] + bias[0][h];
                float pf = acc[1 * 2 + h][r] + bias[1][h];
                float po = acc[2 * 2 + h][r] + bias[2][h];
                float pz = acc[3 * 2 + h][r] + bias[3][h];
                float ig = fast_sigmoid(pi);
                float fg = fast_sigmoid(pf);
                float og = fast_sigmoid(po);
                float zg = fast_tanh(pz);
                float cn = ig * zg + fg * cst[q];
                cst[q] = cn;
                float hn = og * fast_tanh(cn);
                hsum[q] += mk[r] * hn;
                xh[(8 * hi + r) * XH_STRIDE + KX + j] = f2bf(hn);   // h for next step
            }
        }
        // top-of-loop barrier A covers the h-writeback -> next matmul dependency
    }

    // ---- masked time-average output ----
    #pragma unroll
    for (int h = 0; h < 2; ++h) {
        int j = wave * 32 + h * 16 + l16;
        #pragma unroll
        for (int r = 0; r < 8; ++r) {
            out[(size_t)(b0 + 8 * hi + r) * H_ + j] = hsum[h * 8 + r] / smsum[r];
        }
    }
}

extern "C" void kernel_launch(void* const* d_in, const int* in_sizes, int n_in,
                              void* d_out, int out_size, void* d_ws, size_t ws_size,
                              hipStream_t stream) {
    const float* x    = (const float*)d_in[0];
    const float* msk  = (const float*)d_in[1];
    const float* Wi   = (const float*)d_in[2];
    const float* bi   = (const float*)d_in[3];
    const float* Wf   = (const float*)d_in[4];
    const float* bfv  = (const float*)d_in[5];
    const float* Wo   = (const float*)d_in[6];
    const float* bo   = (const float*)d_in[7];
    const float* Wz   = (const float*)d_in[8];
    const float* bz   = (const float*)d_in[9];
    __bf16* Bfrag = (__bf16*)d_ws;           // needs 64*15*32*16*2 = 983,040 bytes
    float* out = (float*)d_out;

    pack_bfrag<<<(NT_COUNT * KSTEPS * 32 + 255) / 256, 256, 0, stream>>>(Wi, Wf, Wo, Wz, Bfrag);
    lstm_persistent<<<16, 256, 0, stream>>>(x, msk, bi, bfv, bo, bz, Bfrag, out);
}
